// Attention_53566832115785
// MI455X (gfx1250) — matrix operand order repaired and checked by reading, not compiled, not run on hardware
//
#include <hip/hip_runtime.h>

typedef _Float16 f16;
typedef __attribute__((ext_vector_type(16))) _Float16 v16h;
typedef __attribute__((ext_vector_type(8)))  _Float16 v8h;
typedef __attribute__((ext_vector_type(8)))  float    v8f;

#define B_  2
#define S_  2048
#define D_  1024
#define H_  16
#define HD_ 64
#define PAIRS_ 32     // HD_/2
#define LPAD_  40     // padded LDS row stride (halves): 80B, 16B-aligned, bank-friendly

__device__ __forceinline__ v16h cat16(v8h lo, v8h hi) {
  return __builtin_shufflevector(lo, hi, 0,1,2,3,4,5,6,7,8,9,10,11,12,13,14,15);
}

__device__ __forceinline__ v8f wmma_f16(v16h a, v16h b, v8f c) {
  // D = A(16x32 f16) * B(32x16 f16) + C(16x16 f32)
  return __builtin_amdgcn_wmma_f32_16x16x32_f16(false, a, false, b, (short)0, c,
                                                false, false);
}

// Async 16B-per-lane copy global -> LDS (CDNA5 GLOBAL_LOAD_ASYNC_TO_LDS_B128,
// tracked by ASYNCcnt). LDS aperture maps addr[31:0] to the LDS byte offset,
// so truncating the generic shared pointer gives the VDST operand.
__device__ __forceinline__ void async_copy_b128(void* lds_dst, const void* gsrc) {
  unsigned lds_addr = (unsigned)(unsigned long long)lds_dst;
  asm volatile("global_load_async_to_lds_b128 %0, %1, off"
               :: "v"(lds_addr), "v"(gsrc) : "memory");
}
__device__ __forceinline__ void async_wait0() {
  asm volatile("s_wait_asynccnt 0x0" ::: "memory");
}

// ---------------------------------------------------------------- convert ---
__global__ __launch_bounds__(256)
void f32_to_f16_kernel(const float* __restrict__ in, f16* __restrict__ out, int n) {
  for (int i = blockIdx.x * blockDim.x + threadIdx.x; i < n;
       i += gridDim.x * blockDim.x)
    out[i] = (f16)in[i];
}

// ------------------------------------------------------------------- GEMM ---
// out(M,N) = A(M,K) @ W(N,K)^T + bias   (A, W f16; f32 accumulate)
// 256 threads = 8 waves as 4(M) x 2(N); block tile 128x128; wave tile 32x64.
// K-slices staged to LDS with async global->LDS copies, double buffered.
template <bool OUT_F16>
__global__ void __launch_bounds__(256)
gemm_bias_wT(const f16* __restrict__ A, const f16* __restrict__ W,
             const float* __restrict__ bias, void* __restrict__ outp,
             int M, int N, int K) {
  __shared__ f16 ldsA[2][128][LPAD_];   // 20 KB
  __shared__ f16 ldsB[2][128][LPAD_];   // 20 KB

  const int t    = threadIdx.x;
  const int lane = t & 31, wave = t >> 5;
  const int ln   = lane & 15, lh = lane >> 4;
  const int mw   = wave >> 1, nw = wave & 1;   // wave position in 4x2 grid
  const int mblk = blockIdx.y * 128;
  const int nblk = blockIdx.x * 128;
  const int koff = lh * 8;                     // A-fragment K sub-offset
  const int kb2  = lh * 8;                     // B-fragment K sub-offset

  // each thread stages 2x16B chunks per tile per k-step (512 chunks/tile)
  auto stage = [&](int buf, int k0) {
#pragma unroll
    for (int c = 0; c < 2; ++c) {
      int chunk = t + c * 256;                 // 0..511
      int row   = chunk >> 2;                  // 0..127
      int off   = (chunk & 3) * 8;             // 0,8,16,24
      async_copy_b128(&ldsA[buf][row][off],
                      A + (size_t)(mblk + row) * K + k0 + off);
      async_copy_b128(&ldsB[buf][row][off],
                      W + (size_t)(nblk + row) * K + k0 + off);
    }
  };

  v8f acc[2][4] = {};
  stage(0, 0);
  for (int k0 = 0; k0 < K; k0 += 32) {
    const int buf = (k0 >> 5) & 1;
    async_wait0();                             // our async writes landed in LDS
    __syncthreads();                           // whole block's staging visible
    if (k0 + 32 < K) stage(buf ^ 1, k0 + 32);  // prefetch next slice async

    // A fragments for the wave's two 16-row subtiles (ISA 16-bit A 16x32 layout)
    v16h af[2];
#pragma unroll
    for (int ms = 0; ms < 2; ++ms) {
      const f16* ar = &ldsA[buf][mw * 32 + ms * 16 + ln][0];
      af[ms] = cat16(*(const v8h*)(ar + koff), *(const v8h*)(ar + 16 + koff));
    }
    // 4 B fragments x 2 A fragments = 8 WMMAs per k-step
#pragma unroll
    for (int nt = 0; nt < 4; ++nt) {
      const f16* br = &ldsB[buf][nw * 64 + nt * 16 + ln][0];
      v16h bf = cat16(*(const v8h*)(br + kb2), *(const v8h*)(br + 16 + kb2));
      acc[0][nt] = wmma_f16(af[0], bf, acc[0][nt]);
      acc[1][nt] = wmma_f16(af[1], bf, acc[1][nt]);
    }
  }

  // Epilogue: C layout -> VGPR v: lanes0-15 row base+v, lanes16-31 row base+v+8
#pragma unroll
  for (int ms = 0; ms < 2; ++ms) {
    int mbase = mblk + mw * 32 + ms * 16;
#pragma unroll
    for (int nt = 0; nt < 4; ++nt) {
      int   col = nblk + nw * 64 + nt * 16 + ln;
      float bv  = bias[col];
#pragma unroll
      for (int v = 0; v < 8; ++v) {
        int   row = mbase + v + lh * 8;
        float val = acc[ms][nt][v] + bv;
        if (OUT_F16) ((f16*)outp)[(size_t)row * N + col] = (f16)val;
        else         ((float*)outp)[(size_t)row * N + col] = val;
      }
    }
  }
}

// -------------------------------------------------------- flash attention ---
// One block = one (b, h, 128-query chunk). 8 waves, each owns 16 queries.
// Online softmax, 32 keys per step (PV is a full K=32 WMMA).
__global__ void __launch_bounds__(256)
flash_attn(const f16* __restrict__ qh, const f16* __restrict__ kh,
           const f16* __restrict__ vh, const float* __restrict__ pcos,
           const float* __restrict__ psin, f16* __restrict__ oh) {
  __shared__ f16 ldsQ[128][HD_];     // 16 KB, RoPE'd + scaled Q
  __shared__ f16 ldsK[32][HD_];      // 4 KB, RoPE'd K rows
  __shared__ f16 ldsVt[HD_][32];     // 4 KB, V transposed [dim][key]
  __shared__ f16 ldsP[8][16][32];    // 8 KB, per-wave prob tile (re-layout)

  const int qb0  = blockIdx.x * 128;
  const int h    = blockIdx.y;
  const int b    = blockIdx.z;
  const int t    = threadIdx.x;
  const int lane = t & 31, wave = t >> 5;
  const int ln   = lane & 15, lh = lane >> 4;
  const int kb2v = lh * 8;

  // ---- stage Q once: RoPE + fold in softmax scale 1/sqrt(64) = 0.125
  for (int i = 0; i < 16; ++i) {
    int pi = t * 16 + i;                 // 4096 pairs
    int r = pi >> 5, p = pi & 31;
    int sg = qb0 + r;
    const f16* qrow = qh + ((size_t)(b * S_ + sg) * D_ + h * HD_);
    float c = pcos[sg * PAIRS_ + p], s = psin[sg * PAIRS_ + p];
    float a = (float)qrow[2 * p], bq = (float)qrow[2 * p + 1];
    ldsQ[r][2 * p]     = (f16)((a * c - bq * s) * 0.125f);
    ldsQ[r][2 * p + 1] = (f16)((a * s + bq * c) * 0.125f);
  }
  __syncthreads();

  const int qbase = qb0 + wave * 16;
  const int qhiW  = qbase + 15;

  // ---- Q A-fragments (invariant over key loop)
  v16h aQ0, aQ1;
  {
    const f16* qr = &ldsQ[wave * 16 + ln][0];
    aQ0 = cat16(*(const v8h*)(qr + 0  + lh * 8), *(const v8h*)(qr + 16 + lh * 8));
    aQ1 = cat16(*(const v8h*)(qr + 32 + lh * 8), *(const v8h*)(qr + 48 + lh * 8));
  }

  v8f   acc[4] = {};
  float m_i[8], l_i[8];
#pragma unroll
  for (int v = 0; v < 8; ++v) { m_i[v] = -1e30f; l_i[v] = 0.f; }

  const int nkb = ((qb0 + 127) >> 5) + 1;   // causal: keys up to block's max q
  for (int kb = 0; kb < nkb; ++kb) {
    const int kb0 = kb * 32;
    __syncthreads();                         // previous step's K/V consumed
    // ---- stage K (RoPE, no scale)
    for (int i = 0; i < 4; ++i) {
      int pi = t * 4 + i;                    // 1024 pairs
      int r = pi >> 5, p = pi & 31;
      int kg = kb0 + r;
      const f16* krow = kh + ((size_t)(b * S_ + kg) * D_ + h * HD_);
      float c = pcos[kg * PAIRS_ + p], s = psin[kg * PAIRS_ + p];
      float a = (float)krow[2 * p], bk = (float)krow[2 * p + 1];
      ldsK[r][2 * p]     = (f16)(a * c - bk * s);
      ldsK[r][2 * p + 1] = (f16)(a * s + bk * c);
    }
    // ---- stage V transposed: one 16B global load/thread, scatter to LDS
    {
      int key   = t >> 3;                    // 0..31
      int dbase = (t & 7) * 8;               // 0..56
      v8h vv = *(const v8h*)(vh + (size_t)(b * S_ + kb0 + key) * D_ +
                             h * HD_ + dbase);
#pragma unroll
      for (int j = 0; j < 8; ++j) ldsVt[dbase + j][key] = vv[j];
    }
    __syncthreads();

    if (kb0 > qhiW) continue;                // wave-uniform; barriers already done

    // ---- scores: two 16x16 tiles, K=64 via 2 WMMAs each
    v8f sc0 = {}, sc1 = {};
    {
      v16h b0c0 = cat16(*(const v8h*)(&ldsK[ln][kb2v]), *(const v8h*)(&ldsK[ln][16 + kb2v]));
      v16h b0c1 = cat16(*(const v8h*)(&ldsK[ln][32 + kb2v]), *(const v8h*)(&ldsK[ln][48 + kb2v]));
      sc0 = wmma_f16(aQ0, b0c0, sc0);
      sc0 = wmma_f16(aQ1, b0c1, sc0);
      v16h b1c0 = cat16(*(const v8h*)(&ldsK[16 + ln][kb2v]), *(const v8h*)(&ldsK[16 + ln][16 + kb2v]));
      v16h b1c1 = cat16(*(const v8h*)(&ldsK[16 + ln][32 + kb2v]), *(const v8h*)(&ldsK[16 + ln][48 + kb2v]));
      sc1 = wmma_f16(aQ0, b1c0, sc1);
      sc1 = wmma_f16(aQ1, b1c1, sc1);
    }

    // ---- online softmax (rows live in 16-lane halves of the wave)
#pragma unroll
    for (int v = 0; v < 8; ++v) {
      int  row = qbase + v + lh * 8;
      bool ok0 = (kb0 + ln) <= row;
      bool ok1 = (kb0 + 16 + ln) <= row;
      float x0 = ok0 ? sc0[v] : -1e30f;
      float x1 = ok1 ? sc1[v] : -1e30f;
      float mt = fmaxf(x0, x1);
      for (int off = 1; off < 16; off <<= 1)
        mt = fmaxf(mt, __shfl_xor(mt, off, 32));
      float mnew  = fmaxf(m_i[v], mt);
      float alpha = __expf(m_i[v] - mnew);
      float p0 = ok0 ? __expf(x0 - mnew) : 0.f;
      float p1 = ok1 ? __expf(x1 - mnew) : 0.f;
      float ps = p0 + p1;
      for (int off = 1; off < 16; off <<= 1)
        ps += __shfl_xor(ps, off, 32);
      l_i[v] = l_i[v] * alpha + ps;
      m_i[v] = mnew;
#pragma unroll
      for (int nt = 0; nt < 4; ++nt) acc[nt][v] *= alpha;
      int rl = v + lh * 8;
      ldsP[wave][rl][ln]      = (f16)p0;     // C layout -> LDS (re-layout bounce)
      ldsP[wave][rl][16 + ln] = (f16)p1;
    }

    // ---- PV: P(16x32) @ V(32x64), 4 WMMAs
    {
      const f16* pr = &ldsP[wave][ln][0];
      v16h aP = cat16(*(const v8h*)(pr + lh * 8),
                      *(const v8h*)(pr + 16 + lh * 8));
#pragma unroll
      for (int nt = 0; nt < 4; ++nt) {
        v16h bV = cat16(*(const v8h*)(&ldsVt[nt * 16 + ln][kb2v]), *(const v8h*)(&ldsVt[nt * 16 + ln][16 + kb2v]));
        acc[nt] = wmma_f16(aP, bV, acc[nt]);
      }
    }
  }

  // ---- normalize + write attention output (f16, layout [b][s][h*64+d])
#pragma unroll
  for (int v = 0; v < 8; ++v) {
    float inv  = 1.f / l_i[v];
    int   srow = qbase + v + lh * 8;
#pragma unroll
    for (int nt = 0; nt < 4; ++nt) {
      int d = h * HD_ + nt * 16 + ln;
      oh[(size_t)(b * S_ + srow) * D_ + d] = (f16)(acc[nt][v] * inv);
    }
  }
}

// ------------------------------------------------------------------- host ---
extern "C" void kernel_launch(void* const* d_in, const int* in_sizes, int n_in,
                              void* d_out, int out_size, void* d_ws, size_t ws_size,
                              hipStream_t stream) {
  (void)in_sizes; (void)n_in; (void)out_size; (void)ws_size;
  const float* x    = (const float*)d_in[0];
  const float* pcos = (const float*)d_in[1];
  const float* psin = (const float*)d_in[2];
  const float* wq_w = (const float*)d_in[3];
  const float* wq_b = (const float*)d_in[4];
  const float* wk_w = (const float*)d_in[5];
  const float* wk_b = (const float*)d_in[6];
  const float* wv_w = (const float*)d_in[7];
  const float* wv_b = (const float*)d_in[8];
  const float* wo_w = (const float*)d_in[9];
  const float* wo_b = (const float*)d_in[10];

  const size_t BSD = (size_t)B_ * S_ * D_;   // 4,194,304 elems
  const size_t DD  = (size_t)D_ * D_;        // 1,048,576 elems
  char* ws = (char*)d_ws;
  f16* xh  = (f16*)(ws);
  f16* wqh = (f16*)(ws + BSD * 2);
  f16* wkh = (f16*)(ws + BSD * 2 + DD * 2);
  f16* wvh = (f16*)(ws + BSD * 2 + DD * 4);
  f16* woh = (f16*)(ws + BSD * 2 + DD * 6);
  f16* qh  = (f16*)(ws + BSD * 2 + DD * 8);
  f16* kh  = (f16*)(ws + BSD * 4 + DD * 8);
  f16* vh  = (f16*)(ws + BSD * 6 + DD * 8);
  f16* oh  = (f16*)(ws + BSD * 8 + DD * 8);  // total ws use ~50 MB

  f32_to_f16_kernel<<<2048, 256, 0, stream>>>(x, xh, (int)BSD);
  f32_to_f16_kernel<<<1024, 256, 0, stream>>>(wq_w, wqh, (int)DD);
  f32_to_f16_kernel<<<1024, 256, 0, stream>>>(wk_w, wkh, (int)DD);
  f32_to_f16_kernel<<<1024, 256, 0, stream>>>(wv_w, wvh, (int)DD);
  f32_to_f16_kernel<<<1024, 256, 0, stream>>>(wo_w, woh, (int)DD);

  dim3 gg(D_ / 128, (B_ * S_) / 128);        // 8 x 32 blocks
  gemm_bias_wT<true ><<<gg, 256, 0, stream>>>(xh, wqh, wq_b, qh, B_ * S_, D_, D_);
  gemm_bias_wT<true ><<<gg, 256, 0, stream>>>(xh, wkh, wk_b, kh, B_ * S_, D_, D_);
  gemm_bias_wT<true ><<<gg, 256, 0, stream>>>(xh, wvh, wv_b, vh, B_ * S_, D_, D_);

  dim3 ga(S_ / 128, H_, B_);                 // 16 x 16 x 2 blocks
  flash_attn<<<ga, 256, 0, stream>>>(qh, kh, vh, pcos, psin, oh);

  gemm_bias_wT<false><<<gg, 256, 0, stream>>>(oh, woh, wo_b, d_out, B_ * S_, D_, D_);
}
